// GNNConvLayer_55929064129202
// MI455X (gfx1250) — compile-verified, matrix operand order checked by reading, not run on hardware
//
#include <hip/hip_runtime.h>
#include <hip/hip_bf16.h>

// ---------------------------------------------------------------------------
// GNN conv layer for MI455X (gfx1250, wave32, WMMA bf16 -> f32 accum)
// ---------------------------------------------------------------------------

#define D_   128
#define DE_  32
#define FIN_ 288            // 2*D + DE
#define RS_  288            // feats row stride (elements)
#define HID_ 512
#define EPS_ 1e-5f

typedef __attribute__((ext_vector_type(16))) __bf16 v16bf;
typedef __attribute__((ext_vector_type(8)))  float  v8f;

#define LDS_FENCE() asm volatile("s_wait_dscnt 0x0" ::: "memory")

static __device__ __forceinline__ unsigned pack2bf(float a, float b) {
    union { __bf16 h[2]; unsigned u; } x;
    x.h[0] = (__bf16)a; x.h[1] = (__bf16)b;
    return x.u;
}

static __device__ __forceinline__ v16bf frag_from(uint4 lo, uint4 hi) {
    union { uint4 u[2]; v16bf v; } x;
    x.u[0] = lo; x.u[1] = hi;
    return x.v;
}

// A-fragment (16x32 bf16, M=row) from row-major bf16 LDS matrix, row stride rs.
// Lane l<16: row=l, K = kc+{0..7,16..23}; lane>=16: row=l-16, K = kc+{8..15,24..31}.
static __device__ __forceinline__ v16bf load_a_bf16(const __bf16* m, int rs, int kc, int lane) {
    int r  = lane & 15;
    int kb = (lane & 16) ? 8 : 0;
    const __bf16* p = m + r * rs + kc + kb;
    uint4 lo = *(const uint4*)(p);
    uint4 hi = *(const uint4*)(p + 16);
    return frag_from(lo, hi);
}

// Same A-fragment but source rows are f32 in LDS (stride 128), converted on load.
static __device__ __forceinline__ v16bf load_a_f32(const float* m, int kc, int lane) {
    int r  = lane & 15;
    int kb = (lane & 16) ? 8 : 0;
    const float* p = m + r * 128 + kc + kb;
    float4 a0 = *(const float4*)(p);
    float4 a1 = *(const float4*)(p + 4);
    float4 a2 = *(const float4*)(p + 16);
    float4 a3 = *(const float4*)(p + 20);
    uint4 lo, hi;
    lo.x = pack2bf(a0.x, a0.y); lo.y = pack2bf(a0.z, a0.w);
    lo.z = pack2bf(a1.x, a1.y); lo.w = pack2bf(a1.z, a1.w);
    hi.x = pack2bf(a2.x, a2.y); hi.y = pack2bf(a2.z, a2.w);
    hi.z = pack2bf(a3.x, a3.y); hi.w = pack2bf(a3.z, a3.w);
    return frag_from(lo, hi);
}

// B-fragment (32x16 bf16, N=column) from LDS weights stored col-major:
// bt[col*K + k].  colBase = first column of the 16-wide tile.
static __device__ __forceinline__ v16bf load_b_bf16(const __bf16* bt, int K, int colBase,
                                                    int kc, int lane) {
    int c  = colBase + (lane & 15);
    int kb = (lane & 16) ? 8 : 0;
    const __bf16* p = bt + c * K + kc + kb;
    uint4 lo = *(const uint4*)(p);
    uint4 hi = *(const uint4*)(p + 16);
    return frag_from(lo, hi);
}

static __device__ __forceinline__ v8f wmma_bf16(v16bf a, v16bf b, v8f c) {
    return __builtin_amdgcn_wmma_f32_16x16x32_bf16(false, a, false, b, (short)0, c,
                                                   false, false);
}

static __device__ __forceinline__ float sigmoidf_(float x) {
    return 1.0f / (1.0f + __expf(-x));
}

// ---------------------------------------------------------------------------
// Kernel 0: zero the workspace (agg[N*128] + cnt[N])
// ---------------------------------------------------------------------------
__global__ void zero_ws_kernel(float* p, long long n) {
    long long i = (long long)blockIdx.x * blockDim.x + threadIdx.x;
    long long stride = (long long)gridDim.x * blockDim.x;
    for (; i < n; i += stride) p[i] = 0.0f;
}

// ---------------------------------------------------------------------------
// Kernel 1: per-edge message MLP + atomic scatter-add into agg / cnt.
// 256 threads = 8 waves; each wave owns a 16-edge tile.
// ---------------------------------------------------------------------------
__global__ __launch_bounds__(256, 1)
void edge_msg_kernel(const float* __restrict__ x, const float* __restrict__ ea,
                     const long long* __restrict__ ei,
                     const float* __restrict__ W1, const float* __restrict__ b1,
                     const float* __restrict__ W2, const float* __restrict__ b2,
                     const float* __restrict__ W3, const float* __restrict__ b3,
                     float* __restrict__ agg, float* __restrict__ cnt,
                     int E_, int N_) {
    extern __shared__ char smem[];
    __bf16* sW1 = (__bf16*)smem;                 // [128][288] col-major (col*288+k)
    __bf16* sW2 = sW1 + 128 * 288;               // [128][128]
    __bf16* sW3 = sW2 + 128 * 128;               // [128][128]
    float*  sb1 = (float*)(sW3 + 128 * 128);     // 128
    float*  sb2 = sb1 + 128;                     // 128
    float*  sb3 = sb2 + 128;                     // 128
    __bf16* sFeat = (__bf16*)(sb3 + 128);        // 8 waves * [16][288]

    const int tid = threadIdx.x;

    // Cache weights transposed (col-major) as bf16, biases as f32.
    for (int i = tid; i < 128 * 288; i += blockDim.x) {
        int n = i / 288, k = i - n * 288;
        sW1[i] = (__bf16)W1[k * 128 + n];
    }
    for (int i = tid; i < 128 * 128; i += blockDim.x) {
        int n = i >> 7, k = i & 127;
        sW2[i] = (__bf16)W2[k * 128 + n];
        sW3[i] = (__bf16)W3[k * 128 + n];
    }
    for (int i = tid; i < 128; i += blockDim.x) {
        sb1[i] = b1[i]; sb2[i] = b2[i]; sb3[i] = b3[i];
    }
    __syncthreads();

    const int wave = tid >> 5, lane = tid & 31;
    __bf16* feats = sFeat + wave * (16 * RS_);

    const long long* srcIdx = ei;
    const long long* dstIdx = ei + E_;

    const int tilesTotal = (E_ + 15) >> 4;
    const int tileStep   = gridDim.x * 8;
    const int cb    = lane & 15;
    const bool upper = lane >= 16;

    for (int tile = blockIdx.x * 8 + wave; tile < tilesTotal; tile += tileStep) {
        const int e0 = tile << 4;

        // ---- gather: 2 lanes per edge, each covers 64 features of x ----
        const int j = lane >> 1, half = lane & 1;
        const long long e  = (long long)e0 + j;
        const bool ve      = e < (long long)E_;
        const long long es = ve ? e : (long long)(E_ - 1);
        const long long s  = srcIdx[es];
        const long long d2 = dstIdx[es];

        // dst index for scatter (lane r<16 holds dst of row r)
        long long eL = (long long)e0 + cb;
        if (eL >= E_) eL = E_ - 1;
        const int dL = (int)dstIdx[eL];

        // prefetch next tile's edge_attr rows
        {
            long long enext = (long long)(tile + tileStep) << 4;
            if (enext < E_) __builtin_prefetch(ea + enext * DE_ + lane * 8, 0, 1);
        }

        const float* ps = x + s * D_ + half * 64;
        const float* pd = x + d2 * D_ + half * 64;
        __bf16* frow = feats + j * RS_;
        #pragma unroll
        for (int q = 0; q < 16; ++q) {
            float4 vs = ve ? ((const float4*)ps)[q] : make_float4(0.f, 0.f, 0.f, 0.f);
            float4 vd = ve ? ((const float4*)pd)[q] : make_float4(0.f, 0.f, 0.f, 0.f);
            uint2 us; us.x = pack2bf(vs.x, vs.y); us.y = pack2bf(vs.z, vs.w);
            uint2 ud; ud.x = pack2bf(vd.x, vd.y); ud.y = pack2bf(vd.z, vd.w);
            *(uint2*)(frow + half * 64 + q * 4)       = us;
            *(uint2*)(frow + 128 + half * 64 + q * 4) = ud;
        }
        if (half == 0) {
            const float* pe = ea + es * DE_;
            #pragma unroll
            for (int q = 0; q < 8; ++q) {
                float4 va = ve ? ((const float4*)pe)[q] : make_float4(0.f, 0.f, 0.f, 0.f);
                uint2 u; u.x = pack2bf(va.x, va.y); u.y = pack2bf(va.z, va.w);
                *(uint2*)(frow + 256 + q * 4) = u;
            }
        }
        LDS_FENCE();

        // ---- GEMM1: m1 = feats[16x288] @ W1 + b1 ----
        v8f acc[8];
        #pragma unroll
        for (int t = 0; t < 8; ++t) acc[t] = (v8f){0, 0, 0, 0, 0, 0, 0, 0};
        #pragma unroll
        for (int k0 = 0; k0 < 9; ++k0) {
            v16bf a = load_a_bf16(feats, RS_, k0 * 32, lane);
            #pragma unroll
            for (int t = 0; t < 8; ++t) {
                v16bf b = load_b_bf16(sW1, FIN_, t * 16, k0 * 32, lane);
                acc[t] = wmma_bf16(a, b, acc[t]);
            }
        }
        // + b1, store bf16 back to feats[0..127] as m1 (feats fully consumed)
        #pragma unroll
        for (int t = 0; t < 8; ++t) {
            const int col = t * 16 + cb;
            const float bias = sb1[col];
            #pragma unroll
            for (int v = 0; v < 8; ++v) {
                const int r = upper ? v + 8 : v;
                feats[r * RS_ + col] = (__bf16)(acc[t][v] + bias);
            }
        }
        LDS_FENCE();

        // ---- GEMM2: m2 = sigmoid(m1 @ W2 + b2) ----
        #pragma unroll
        for (int t = 0; t < 8; ++t) acc[t] = (v8f){0, 0, 0, 0, 0, 0, 0, 0};
        #pragma unroll
        for (int k0 = 0; k0 < 4; ++k0) {
            v16bf a = load_a_bf16(feats, RS_, k0 * 32, lane);
            #pragma unroll
            for (int t = 0; t < 8; ++t) {
                v16bf b = load_b_bf16(sW2, 128, t * 16, k0 * 32, lane);
                acc[t] = wmma_bf16(a, b, acc[t]);
            }
        }
        #pragma unroll
        for (int t = 0; t < 8; ++t) {
            const int col = t * 16 + cb;
            const float bias = sb2[col];
            #pragma unroll
            for (int v = 0; v < 8; ++v) {
                const int r = upper ? v + 8 : v;
                feats[r * RS_ + col] = (__bf16)sigmoidf_(acc[t][v] + bias);
            }
        }
        LDS_FENCE();

        // ---- GEMM3: m3 = m2 @ W3 + b3 ----
        #pragma unroll
        for (int t = 0; t < 8; ++t) acc[t] = (v8f){0, 0, 0, 0, 0, 0, 0, 0};
        #pragma unroll
        for (int k0 = 0; k0 < 4; ++k0) {
            v16bf a = load_a_bf16(feats, RS_, k0 * 32, lane);
            #pragma unroll
            for (int t = 0; t < 8; ++t) {
                v16bf b = load_b_bf16(sW3, 128, t * 16, k0 * 32, lane);
                acc[t] = wmma_bf16(a, b, acc[t]);
            }
        }

        // ---- scatter-add rows into agg[dst], count into cnt[dst] ----
        int nodeOf[8];
        #pragma unroll
        for (int v = 0; v < 8; ++v) {
            const int r = upper ? v + 8 : v;
            nodeOf[v] = __shfl(dL, r, 32);
        }
        #pragma unroll
        for (int t = 0; t < 8; ++t) {
            const int col = t * 16 + cb;
            const float bias = sb3[col];
            #pragma unroll
            for (int v = 0; v < 8; ++v) {
                const int r = upper ? v + 8 : v;
                if (e0 + r < E_)
                    atomicAdd(&agg[(long long)nodeOf[v] * D_ + col], acc[t][v] + bias);
            }
        }
        if (lane < 16 && (e0 + lane) < E_) atomicAdd(&cnt[dL], 1.0f);
    }
}

// ---------------------------------------------------------------------------
// Kernel 2: dh = agg/max(cnt,1); h = LN(x+dh); ff = sig(h@Wf1+bf1)@Wf2+bf2;
//           out = LN(h+ff).   128 threads = 4 waves; wave owns 16-node tile.
// ---------------------------------------------------------------------------
__global__ __launch_bounds__(128, 1)
void node_ff_kernel(const float* __restrict__ x, const float* __restrict__ agg,
                    const float* __restrict__ cnt,
                    const float* __restrict__ Wf1, const float* __restrict__ bf1,
                    const float* __restrict__ Wf2, const float* __restrict__ bf2,
                    const float* __restrict__ g0, const float* __restrict__ be0,
                    const float* __restrict__ g1, const float* __restrict__ be1,
                    float* __restrict__ out, int N_) {
    extern __shared__ char smem[];
    __bf16* sWf1 = (__bf16*)smem;                 // [512][128] col-major
    __bf16* sWf2 = sWf1 + HID_ * 128;             // [128][512] col-major
    float*  sbf1 = (float*)(sWf2 + 128 * HID_);   // 512
    float*  sbf2 = sbf1 + HID_;                   // 128
    float*  sg0  = sbf2 + 128;                    // 128
    float*  sbe0 = sg0 + 128;                     // 128
    float*  sg1  = sbe0 + 128;                    // 128
    float*  sbe1 = sg1 + 128;                     // 128
    float*  hAll = (float*)(sbe1 + 128);          // 4 waves * [16][128] f32
    __bf16* sAll = (__bf16*)(hAll + 4 * 16 * 128);// 4 waves * [16][128] bf16

    const int tid = threadIdx.x;

    for (int i = tid; i < HID_ * 128; i += blockDim.x) {
        int n = i >> 7, k = i & 127;               // Wf1t[n][k] = Wf1[k][n]
        sWf1[i] = (__bf16)Wf1[k * HID_ + n];
    }
    for (int i = tid; i < 128 * HID_; i += blockDim.x) {
        int n = i >> 9, k = i & 511;               // Wf2t[n][k] = Wf2[k][n]
        sWf2[i] = (__bf16)Wf2[k * 128 + n];
    }
    for (int i = tid; i < HID_; i += blockDim.x) sbf1[i] = bf1[i];
    for (int i = tid; i < 128; i += blockDim.x) {
        sbf2[i] = bf2[i]; sg0[i] = g0[i]; sbe0[i] = be0[i];
        sg1[i] = g1[i];  sbe1[i] = be1[i];
    }
    __syncthreads();

    const int wave = tid >> 5, lane = tid & 31;
    float*  hbuf = hAll + wave * (16 * 128);
    __bf16* sbuf = sAll + wave * (16 * 128);

    const int tiles = (N_ + 15) >> 4;
    const int cb    = lane & 15;
    const bool upper = lane >= 16;

    for (int tile = blockIdx.x * 4 + wave; tile < tiles; tile += gridDim.x * 4) {
        const int n0 = tile << 4;

        // ---- phase A: z = x + agg/max(cnt,1); first LayerNorm -> hbuf ----
        const int row  = lane >> 1, half = lane & 1;
        const int node = n0 + row;
        const bool vn  = node < N_;
        const int ns   = vn ? node : N_ - 1;
        const float cv = cnt[ns];
        const float sc = 1.0f / fmaxf(cv, 1.0f);
        const float* px = x   + (long long)ns * D_ + half * 64;
        const float* pa = agg + (long long)ns * D_ + half * 64;

        float4 z[16];
        float sum = 0.0f, ss = 0.0f;
        #pragma unroll
        for (int q = 0; q < 16; ++q) {
            float4 xa = ((const float4*)px)[q];
            float4 ag = ((const float4*)pa)[q];
            z[q].x = xa.x + ag.x * sc; z[q].y = xa.y + ag.y * sc;
            z[q].z = xa.z + ag.z * sc; z[q].w = xa.w + ag.w * sc;
            sum += z[q].x + z[q].y + z[q].z + z[q].w;
            ss  += z[q].x * z[q].x + z[q].y * z[q].y + z[q].z * z[q].z + z[q].w * z[q].w;
        }
        sum += __shfl_xor(sum, 1, 32);
        ss  += __shfl_xor(ss, 1, 32);
        const float mu   = sum * (1.0f / 128.0f);
        const float var  = ss * (1.0f / 128.0f) - mu * mu;
        const float rstd = rsqrtf(var + EPS_);
        float* hrow = hbuf + row * 128 + half * 64;
        #pragma unroll
        for (int q = 0; q < 16; ++q) {
            const int c = half * 64 + q * 4;
            float4 h4;
            h4.x = (z[q].x - mu) * rstd * sg0[c + 0] + sbe0[c + 0];
            h4.y = (z[q].y - mu) * rstd * sg0[c + 1] + sbe0[c + 1];
            h4.z = (z[q].z - mu) * rstd * sg0[c + 2] + sbe0[c + 2];
            h4.w = (z[q].w - mu) * rstd * sg0[c + 3] + sbe0[c + 3];
            ((float4*)hrow)[q] = h4;
        }
        LDS_FENCE();

        // ---- phase B: ff = sigmoid(h@Wf1+bf1) @ Wf2, panel-blocked over HID ----
        v8f ffacc[8];
        #pragma unroll
        for (int t = 0; t < 8; ++t) ffacc[t] = (v8f){0, 0, 0, 0, 0, 0, 0, 0};

        for (int p = 0; p < 4; ++p) {
            v8f sacc[8];
            #pragma unroll
            for (int t = 0; t < 8; ++t) sacc[t] = (v8f){0, 0, 0, 0, 0, 0, 0, 0};
            #pragma unroll
            for (int k0 = 0; k0 < 4; ++k0) {
                v16bf a = load_a_f32(hbuf, k0 * 32, lane);
                #pragma unroll
                for (int t = 0; t < 8; ++t) {
                    v16bf b = load_b_bf16(sWf1, 128, p * 128 + t * 16, k0 * 32, lane);
                    sacc[t] = wmma_bf16(a, b, sacc[t]);
                }
            }
            // bias + sigmoid -> bf16 panel buffer
            #pragma unroll
            for (int t = 0; t < 8; ++t) {
                const int col = t * 16 + cb;
                const float bias = sbf1[p * 128 + col];
                #pragma unroll
                for (int v = 0; v < 8; ++v) {
                    const int r = upper ? v + 8 : v;
                    sbuf[r * 128 + col] = (__bf16)sigmoidf_(sacc[t][v] + bias);
                }
            }
            LDS_FENCE();
            #pragma unroll
            for (int k0 = 0; k0 < 4; ++k0) {
                v16bf a = load_a_bf16(sbuf, 128, k0 * 32, lane);
                #pragma unroll
                for (int t = 0; t < 8; ++t) {
                    v16bf b = load_b_bf16(sWf2, HID_, t * 16, p * 128 + k0 * 32, lane);
                    ffacc[t] = wmma_bf16(a, b, ffacc[t]);
                }
            }
            LDS_FENCE();
        }

        // ---- phase C: y = h + ff + bf2 (in-place in hbuf), second LN, store ----
        #pragma unroll
        for (int t = 0; t < 8; ++t) {
            const int col = t * 16 + cb;
            const float bias = sbf2[col];
            #pragma unroll
            for (int v = 0; v < 8; ++v) {
                const int r = upper ? v + 8 : v;
                const int idx = r * 128 + col;
                hbuf[idx] = hbuf[idx] + ffacc[t][v] + bias;
            }
        }
        LDS_FENCE();

        float sum2 = 0.0f, ss2 = 0.0f;
        float4 y[16];
        #pragma unroll
        for (int q = 0; q < 16; ++q) {
            y[q] = ((const float4*)hrow)[q];
            sum2 += y[q].x + y[q].y + y[q].z + y[q].w;
            ss2  += y[q].x * y[q].x + y[q].y * y[q].y + y[q].z * y[q].z + y[q].w * y[q].w;
        }
        sum2 += __shfl_xor(sum2, 1, 32);
        ss2  += __shfl_xor(ss2, 1, 32);
        const float mu2   = sum2 * (1.0f / 128.0f);
        const float var2  = ss2 * (1.0f / 128.0f) - mu2 * mu2;
        const float rstd2 = rsqrtf(var2 + EPS_);
        if (vn) {
            float* po = out + (long long)node * D_ + half * 64;
            #pragma unroll
            for (int q = 0; q < 16; ++q) {
                const int c = half * 64 + q * 4;
                float4 o4;
                o4.x = (y[q].x - mu2) * rstd2 * sg1[c + 0] + sbe1[c + 0];
                o4.y = (y[q].y - mu2) * rstd2 * sg1[c + 1] + sbe1[c + 1];
                o4.z = (y[q].z - mu2) * rstd2 * sg1[c + 2] + sbe1[c + 2];
                o4.w = (y[q].w - mu2) * rstd2 * sg1[c + 3] + sbe1[c + 3];
                ((float4*)po)[q] = o4;
            }
        }
    }
}

// ---------------------------------------------------------------------------
// Launcher
// ---------------------------------------------------------------------------
extern "C" void kernel_launch(void* const* d_in, const int* in_sizes, int n_in,
                              void* d_out, int out_size, void* d_ws, size_t ws_size,
                              hipStream_t stream) {
    const float*     x    = (const float*)d_in[0];
    const float*     ea   = (const float*)d_in[1];
    const long long* ei   = (const long long*)d_in[2];
    const float*     W1   = (const float*)d_in[3];
    const float*     b1   = (const float*)d_in[4];
    const float*     W2   = (const float*)d_in[5];
    const float*     b2   = (const float*)d_in[6];
    const float*     W3   = (const float*)d_in[7];
    const float*     b3   = (const float*)d_in[8];
    const float*     Wf1  = (const float*)d_in[9];
    const float*     bf1  = (const float*)d_in[10];
    const float*     Wf2  = (const float*)d_in[11];
    const float*     bf2  = (const float*)d_in[12];
    const float*     g0   = (const float*)d_in[13];
    const float*     be0  = (const float*)d_in[14];
    const float*     g1   = (const float*)d_in[15];
    const float*     be1  = (const float*)d_in[16];
    float* out = (float*)d_out;

    const int N = in_sizes[0] / D_;
    const int E = in_sizes[2] / 2;

    float* agg = (float*)d_ws;             // N * 128
    float* cnt = agg + (size_t)N * D_;     // N

    // dynamic LDS sizes
    const size_t EDGE_SMEM = (size_t)(128 * 288 + 128 * 128 + 128 * 128) * 2  // bf16 W
                           + 3 * 128 * 4                                      // biases
                           + (size_t)8 * 16 * RS_ * 2;                        // feats
    const size_t NODE_SMEM = (size_t)(HID_ * 128 + 128 * HID_) * 2            // bf16 W
                           + (HID_ + 5 * 128) * 4                             // bias/ln
                           + (size_t)4 * 16 * 128 * 4                         // h (f32)
                           + (size_t)4 * 16 * 128 * 2;                        // s (bf16)

    (void)hipFuncSetAttribute(reinterpret_cast<const void*>(edge_msg_kernel),
                              hipFuncAttributeMaxDynamicSharedMemorySize,
                              (int)EDGE_SMEM);
    (void)hipFuncSetAttribute(reinterpret_cast<const void*>(node_ff_kernel),
                              hipFuncAttributeMaxDynamicSharedMemorySize,
                              (int)NODE_SMEM);

    // 0) zero agg + cnt
    {
        long long nz = (long long)N * D_ + N;
        zero_ws_kernel<<<1024, 256, 0, stream>>>((float*)d_ws, nz);
    }
    // 1) edge messages + scatter
    edge_msg_kernel<<<2048, 256, EDGE_SMEM, stream>>>(x, ea, ei, W1, b1, W2, b2, W3, b3,
                                                      agg, cnt, E, N);
    // 2) node aggregate, LN, feed-forward, LN
    node_ff_kernel<<<512, 128, NODE_SMEM, stream>>>(x, agg, cnt, Wf1, bf1, Wf2, bf2,
                                                    g0, be0, g1, be1, out, N);
}
